// TDNN_47571057771215
// MI455X (gfx1250) — 8-bit matrix operand order repaired and checked by reading, compiled, not run on hardware
//
#include <hip/hip_runtime.h>
#include <stdint.h>

typedef int v8i __attribute__((ext_vector_type(8)));

#define L_CTX 3
#define D_IN  1024
#define N_IN  16386
#define T_OUT 16384
#define O_OUT 2048
#define K_DIM 3072

#define BM 128
#define BN 128
#define BK 128
#define NT (K_DIM / BK)
#define LDA 144

#define OFF_PARAMS 0u
#define OFF_PMIN   256u
#define OFF_PMAX   2304u
#define OFF_WSUM   4352u
#define OFF_XSUM   12544u
#define OFF_X3     78336u
#define OFF_QW     144128u
#define OFF_QX     6435840u

__global__ void k_minmax_partial(const float* __restrict__ W,
                                 float* __restrict__ pmin,
                                 float* __restrict__ pmax) {
    __shared__ float smin[256];
    __shared__ float smax[256];
    const int tid = threadIdx.x;
    float lmin =  3.4e38f;
    float lmax = -3.4e38f;
    for (int i = blockIdx.x * 256 + tid; i < O_OUT * K_DIM; i += gridDim.x * 256) {
        float v = W[i];
        lmin = fminf(lmin, v);
        lmax = fmaxf(lmax, v);
    }
    smin[tid] = lmin; smax[tid] = lmax;
    __syncthreads();
    for (int s = 128; s > 0; s >>= 1) {
        if (tid < s) {
            smin[tid] = fminf(smin[tid], smin[tid + s]);
            smax[tid] = fmaxf(smax[tid], smax[tid + s]);
        }
        __syncthreads();
    }
    if (tid == 0) { pmin[blockIdx.x] = smin[0]; pmax[blockIdx.x] = smax[0]; }
}

__global__ void k_params(const float* __restrict__ pmin,
                         const float* __restrict__ pmax,
                         const float* __restrict__ imin_p,
                         const float* __restrict__ imax_p,
                         float* __restrict__ pf,
                         int*   __restrict__ pi) {
    __shared__ float smin[256];
    __shared__ float smax[256];
    const int tid = threadIdx.x;
    smin[tid] = fminf(pmin[tid], pmin[tid + 256]);
    smax[tid] = fmaxf(pmax[tid], pmax[tid + 256]);
    __syncthreads();
    for (int s = 128; s > 0; s >>= 1) {
        if (tid < s) {
            smin[tid] = fminf(smin[tid], smin[tid + s]);
            smax[tid] = fmaxf(smax[tid], smax[tid + s]);
        }
        __syncthreads();
    }
    if (tid == 0) {
        float wmin = smin[0], wmax = smax[0];
        float wsc  = (wmax - wmin) / 254.0f;
        float wzpf = fminf(fmaxf(-127.0f - wmin / wsc, -127.0f), 127.0f);
        int   wzp  = (int)truncf(wzpf);
        float imin = *imin_p, imax = *imax_p;
        float isc  = (imax - imin) / 254.0f;
        float izpf = fminf(fmaxf(-127.0f - imin / isc, -127.0f), 127.0f);
        int   izp  = (int)truncf(izpf);
        pf[0] = wsc;
        pf[1] = isc;
        pf[2] = wsc * isc;
        pi[4] = wzp;
        pi[5] = izp;
        pi[6] = K_DIM * wzp * izp;
    }
}

__global__ void k_quant_weight(const float* __restrict__ W,
                               const float* __restrict__ bias,
                               const float* __restrict__ pf,
                               const int*   __restrict__ pi,
                               signed char* __restrict__ qw,
                               float* __restrict__ dequant,
                               float* __restrict__ out_bias,
                               int*   __restrict__ wsum) {
    const int o   = blockIdx.x;
    const int tid = threadIdx.x;
    const float wsc  = pf[0];
    const float wzpf = (float)pi[4];
    const int   base = o * K_DIM;
    int s = 0;
    for (int k = tid; k < K_DIM; k += 256) {
        float w = W[base + k];
        int   q = (int)rintf(w / wsc + wzpf);
        qw[base + k]      = (signed char)q;
        dequant[base + k] = ((float)q - wzpf) * wsc;
        s += q;
    }
    __shared__ int ssum[256];
    ssum[tid] = s;
    __syncthreads();
    for (int st = 128; st > 0; st >>= 1) {
        if (tid < st) ssum[tid] += ssum[tid + st];
        __syncthreads();
    }
    if (tid == 0) {
        wsum[o]     = ssum[0];
        out_bias[o] = bias[o];
    }
}

__global__ void k_quant_input(const float* __restrict__ X,
                              const float* __restrict__ pf,
                              const int*   __restrict__ pi,
                              signed char* __restrict__ qx,
                              int*   __restrict__ xsum) {
    const int n   = blockIdx.x;
    const int tid = threadIdx.x;
    const float isc  = pf[1];
    const float izpf = (float)pi[5];
    const int base = n * D_IN;
    int s = 0;
    for (int d = tid; d < D_IN; d += 256) {
        float v = X[base + d];
        int   q = (int)rintf(v / isc + izpf);
        qx[base + d] = (signed char)q;
        s += q;
    }
    __shared__ int ssum[256];
    ssum[tid] = s;
    __syncthreads();
    for (int st = 128; st > 0; st >>= 1) {
        if (tid < st) ssum[tid] += ssum[tid + st];
        __syncthreads();
    }
    if (tid == 0) xsum[n] = ssum[0];
}

__global__ void k_x3(const int* __restrict__ xsum,
                     const int* __restrict__ pi,
                     int* __restrict__ x3) {
    const int t = blockIdx.x * 256 + threadIdx.x;
    if (t < T_OUT) x3[t] = pi[4] * (xsum[t] + xsum[t + 1] + xsum[t + 2]);
}

__device__ __forceinline__ void async_copy16(unsigned lds_addr,
                                             unsigned long long gbase,
                                             int goff) {
    asm volatile("global_load_async_to_lds_b128 %0, %1, %2"
                 :
                 : "v"(lds_addr), "v"(goff), "s"(gbase)
                 : "memory");
}

__device__ __forceinline__ void wait_async0() {
    asm volatile("s_wait_asynccnt 0" ::: "memory");
}

__device__ __forceinline__ v8i load_a_frag(const unsigned char* p) {
    int4 a0 = *(const int4*)(p +  0);
    int4 a1 = *(const int4*)(p + 32);
    v8i a;
    a[0] = a0.x; a[1] = a0.y;
    a[2] = a0.z; a[3] = a0.w;
    a[4] = a1.x; a[5] = a1.y;
    a[6] = a1.z; a[7] = a1.w;
    return a;
}

__device__ __forceinline__ v8i load_b_frag(const unsigned char* p) {
    int4 q0 = *(const int4*)(p +  0);
    int4 q1 = *(const int4*)(p + 32);
    v8i b;
    b[0] = q0.x; b[1] = q0.y; b[2] = q0.z; b[3] = q0.w;
    b[4] = q1.x; b[5] = q1.y; b[6] = q1.z; b[7] = q1.w;
    return b;
}

__global__ void __launch_bounds__(256) k_gemm_iu8(
        const signed char* __restrict__ qx,
        const signed char* __restrict__ qw,
        const int*   __restrict__ wsum,
        const int*   __restrict__ x3,
        const float* __restrict__ bias,
        const float* __restrict__ pf,
        const int*   __restrict__ pi,
        float* __restrict__ out1) {
    __shared__ __align__(16) unsigned char sA[2][BM * LDA];
    __shared__ __align__(16) unsigned char sB[2][BN * LDA];

    const int tid  = threadIdx.x;
    const int wave = tid >> 5;
    const int lane = tid & 31;
    const int g    = lane >> 4;
    const int n    = lane & 15;
    const int wm   = wave >> 1;
    const int wn   = wave & 1;

    const int t_blk = blockIdx.x * BM;
    const int o_blk = blockIdx.y * BN;

    const int srow = tid >> 1;
    const int scol = (tid & 1) * 64;
    const unsigned long long gA = (unsigned long long)qx;
    const unsigned long long gB = (unsigned long long)qw;
    const int goffA = (t_blk + srow) * D_IN  + scol;
    const int goffB = (o_blk + srow) * K_DIM + scol;
    unsigned ldsA[2], ldsB[2];
    ldsA[0] = (unsigned)(size_t)&sA[0][srow * LDA + scol];
    ldsA[1] = (unsigned)(size_t)&sA[1][srow * LDA + scol];
    ldsB[0] = (unsigned)(size_t)&sB[0][srow * LDA + scol];
    ldsB[1] = (unsigned)(size_t)&sB[1][srow * LDA + scol];

    const int aoff0 = (wm * 32 +  0 + n) * LDA + g * 16;
    const int aoff1 = (wm * 32 + 16 + n) * LDA + g * 16;
    const int boff0 = (wn * 64 +  0 + n) * LDA + g * 16;
    const int boff1 = (wn * 64 + 16 + n) * LDA + g * 16;
    const int boff2 = (wn * 64 + 32 + n) * LDA + g * 16;
    const int boff3 = (wn * 64 + 48 + n) * LDA + g * 16;

    v8i c[2][4];
#pragma unroll
    for (int mi = 0; mi < 2; ++mi)
#pragma unroll
        for (int j = 0; j < 4; ++j)
#pragma unroll
            for (int r = 0; r < 8; ++r) c[mi][j][r] = 0;

#pragma unroll
    for (int i = 0; i < 4; ++i) {
        async_copy16(ldsA[0] + 16 * i, gA, goffA + 16 * i);
        async_copy16(ldsB[0] + 16 * i, gB, goffB + 16 * i);
    }

    for (int it = 0; it < NT; ++it) {
        const int buf = it & 1;
        wait_async0();
        __syncthreads();

        if (it + 1 < NT) {
            const int knext = (it + 1) * BK;
#pragma unroll
            for (int i = 0; i < 4; ++i) {
                async_copy16(ldsA[buf ^ 1] + 16 * i, gA, goffA + knext + 16 * i);
                async_copy16(ldsB[buf ^ 1] + 16 * i, gB, goffB + knext + 16 * i);
            }
        }

        const unsigned char* bufA = sA[buf];
        const unsigned char* bufB = sB[buf];
#pragma unroll
        for (int kc = 0; kc < BK; kc += 64) {
            v8i a0 = load_a_frag(bufA + aoff0 + kc);
            v8i a1 = load_a_frag(bufA + aoff1 + kc);
            v8i b0 = load_b_frag(bufB + boff0 + kc);
            v8i b1 = load_b_frag(bufB + boff1 + kc);
            v8i b2 = load_b_frag(bufB + boff2 + kc);
            v8i b3 = load_b_frag(bufB + boff3 + kc);

            c[0][0] = __builtin_amdgcn_wmma_i32_16x16x64_iu8(true, a0, true, b0, c[0][0], false, false);
            c[0][1] = __builtin_amdgcn_wmma_i32_16x16x64_iu8(true, a0, true, b1, c[0][1], false, false);
            c[0][2] = __builtin_amdgcn_wmma_i32_16x16x64_iu8(true, a0, true, b2, c[0][2], false, false);
            c[0][3] = __builtin_amdgcn_wmma_i32_16x16x64_iu8(true, a0, true, b3, c[0][3], false, false);
            c[1][0] = __builtin_amdgcn_wmma_i32_16x16x64_iu8(true, a1, true, b0, c[1][0], false, false);
            c[1][1] = __builtin_amdgcn_wmma_i32_16x16x64_iu8(true, a1, true, b1, c[1][1], false, false);
            c[1][2] = __builtin_amdgcn_wmma_i32_16x16x64_iu8(true, a1, true, b2, c[1][2], false, false);
            c[1][3] = __builtin_amdgcn_wmma_i32_16x16x64_iu8(true, a1, true, b3, c[1][3], false, false);
        }
    }

    const float alpha = pf[2];
    const int   izp   = pi[5];
    const int   corr  = pi[6];

#pragma unroll
    for (int mi = 0; mi < 2; ++mi) {
        const int t0 = t_blk + wm * 32 + mi * 16 + 8 * g;
        int x3v[8];
#pragma unroll
        for (int r = 0; r < 8; ++r) x3v[r] = x3[t0 + r];
#pragma unroll
        for (int j = 0; j < 4; ++j) {
            const int   o   = o_blk + wn * 64 + j * 16 + n;
            const int   x2v = izp * wsum[o];
            const float bo  = bias[o];
#pragma unroll
            for (int r = 0; r < 8; ++r) {
                const int v = c[mi][j][r] - x2v - x3v[r] + corr;
                out1[(size_t)(t0 + r) * O_OUT + o] = (float)v * alpha + bo;
            }
        }
    }
}

extern "C" void kernel_launch(void* const* d_in, const int* in_sizes, int n_in,
                              void* d_out, int out_size, void* d_ws, size_t ws_size,
                              hipStream_t stream) {
    (void)in_sizes; (void)n_in; (void)out_size; (void)ws_size;

    const float* x      = (const float*)d_in[0];
    const float* W      = (const float*)d_in[1];
    const float* bias   = (const float*)d_in[2];
    const float* imin_p = (const float*)d_in[3];
    const float* imax_p = (const float*)d_in[4];

    float* out1        = (float*)d_out;
    float* out_dequant = out1 + (size_t)T_OUT * O_OUT;
    float* out_bias    = out_dequant + (size_t)O_OUT * K_DIM;

    char* ws = (char*)d_ws;
    float*       pf   = (float*)(ws + OFF_PARAMS);
    int*         pi   = (int*)(ws + OFF_PARAMS);
    float*       pmin = (float*)(ws + OFF_PMIN);
    float*       pmax = (float*)(ws + OFF_PMAX);
    int*         wsum = (int*)(ws + OFF_WSUM);
    int*         xsum = (int*)(ws + OFF_XSUM);
    int*         x3   = (int*)(ws + OFF_X3);
    signed char* qw   = (signed char*)(ws + OFF_QW);
    signed char* qx   = (signed char*)(ws + OFF_QX);

    k_minmax_partial<<<512, 256, 0, stream>>>(W, pmin, pmax);
    k_params<<<1, 256, 0, stream>>>(pmin, pmax, imin_p, imax_p, pf, pi);
    k_quant_weight<<<O_OUT, 256, 0, stream>>>(W, bias, pf, pi, qw, out_dequant, out_bias, wsum);
    k_quant_input<<<N_IN, 256, 0, stream>>>(x, pf, pi, qx, xsum);
    k_x3<<<(T_OUT + 255) / 256, 256, 0, stream>>>(xsum, pi, x3);

    dim3 grid(T_OUT / BM, O_OUT / BN);
    k_gemm_iu8<<<grid, 256, 0, stream>>>(qx, qw, wsum, x3, bias, pf, pi, out1);
}
